// SwinTransformerBlock_37555194036455
// MI455X (gfx1250) — compile-verified, matrix operand order checked by reading, not run on hardware
//
#include <hip/hip_runtime.h>
#include <hip/hip_bf16.h>

typedef __attribute__((ext_vector_type(16))) _Float16 v16h;
typedef __attribute__((ext_vector_type(8)))  _Float16 v8h;
typedef __attribute__((ext_vector_type(8)))  float    v8f;

#define NB   64      // batch
#define HW   3136    // 56*56
#define CDIM 96
#define NHEAD 3
#define HD   32
#define NTOK 49      // tokens per window
#define NPAD 64      // padded tokens
#define NWIN 64      // windows per image
#define MLPD 384
#define SHIFTV 3
#define QSCALE 0.17677669529663687f  // 1/sqrt(32)
#define NEGBIG -30000.0f

// ---------------- WMMA helpers ----------------

__device__ __forceinline__ v8f wmma_step(v16h a, v16h b, v8f c) {
  // D(f32 16x16) = A(f16 16x32) * B(f16 32x16) + C
  return __builtin_amdgcn_wmma_f32_16x16x32_f16(
      false, a, false, b, (short)0, c, false, false);
}

// Load a 16x32 f16 A-fragment (row-major source, leading dim ld).
// Also used for B: pass W (N x K, row-major) with row0 = n-tile base, since B = W^T.
__device__ __forceinline__ v16h load_frag(const _Float16* __restrict__ base,
                                          int ld, int row0, int col0, int lane) {
  const _Float16* p = base + (row0 + (lane & 15)) * ld + col0 + ((lane >> 4) << 3);
  v8h lo = *(const v8h*)p;
  v8h hi = *(const v8h*)(p + 16);
  v16h r;
#pragma unroll
  for (int j = 0; j < 8; ++j) { r[j] = lo[j]; r[8 + j] = hi[j]; }
  return r;
}

// ---------------- weight fp32 -> f16 ----------------

__global__ void cvt_f32_to_f16(const float* __restrict__ src,
                               _Float16* __restrict__ dst, int n) {
  int i = blockIdx.x * blockDim.x + threadIdx.x;
  if (i < n) dst[i] = (_Float16)src[i];
}

// ---------------- Kernel 1: LN1 + windowed attention + proj + residual ----------------

__global__ __launch_bounds__(256)
void swin_attn_kernel(const float* __restrict__ x,
                      const float* __restrict__ attn_mask,   // (64,49,49)
                      const int*   __restrict__ rel_idx,     // (49,49)
                      const float* __restrict__ qkv_b,       // (288)
                      const float* __restrict__ proj_b,      // (96)
                      const float* __restrict__ rpb,         // (169,3)
                      const float* __restrict__ n1w,
                      const float* __restrict__ n1b,
                      const _Float16* __restrict__ qkv_wh,   // (288,96) f16
                      const _Float16* __restrict__ proj_wh,  // (96,96)  f16
                      float* __restrict__ out) {
  extern __shared__ char smem[];
  _Float16* hA = (_Float16*)smem;    // 64 x 96   (aliased by O after QKV)
  _Float16* qS = hA + NPAD * CDIM;   // 3 x 64 x 32 (row-major per head)
  _Float16* kS = qS + NHEAD * NPAD * HD;
  _Float16* vT = kS + NHEAD * NPAD * HD;   // 3 x 32 x 64 (transposed)
  _Float16* SP = vT + NHEAD * HD * NPAD;   // 64 x 64  logits -> probs (per head, reused)
  _Float16* O  = hA;                       // 64 x 96  attn output (alias)

  const int g  = blockIdx.x;
  const int b  = g >> 6;
  const int nw = g & 63;
  const int wy = nw >> 3, wx = nw & 7;
  const int tid = threadIdx.x;
  const int wave = tid >> 5, lane = tid & 31;

  // ---- Phase 1: LN1 + shifted window gather -> f16 A in LDS ----
  if (tid < NPAD) {
    const int r = tid;
    if (r < NTOK) {
      const int ty = r / 7, tx = r % 7;
      const int sr = (wy * 7 + ty + SHIFTV) % 56;
      const int sc = (wx * 7 + tx + SHIFTV) % 56;
      const float* px = x + ((size_t)b * HW + sr * 56 + sc) * CDIM;
      float mu = 0.f;
      for (int c = 0; c < CDIM; ++c) mu += px[c];
      mu *= (1.0f / CDIM);
      float var = 0.f;
      for (int c = 0; c < CDIM; ++c) { float d = px[c] - mu; var += d * d; }
      var *= (1.0f / CDIM);
      const float inv = rsqrtf(var + 1e-5f);
      for (int c = 0; c < CDIM; ++c)
        hA[r * CDIM + c] = (_Float16)((px[c] - mu) * inv * n1w[c] + n1b[c]);
    } else {
      for (int c = 0; c < CDIM; ++c) hA[r * CDIM + c] = (_Float16)0.0f;
    }
  }
  __syncthreads();

  // ---- Phase 2: QKV = A(64x96) @ W^T(96x288) ----
  // Split into three structurally separate loops (q / k / v) so the
  // destination select is tile-structural, not per-lane -> no EXEC churn.

  // q tiles: output cols [0,96), scaled by 1/sqrt(HD)
  for (int job = wave; job < 24; job += 8) {
    const int mt = job / 6, nt = job % 6;
    v8f acc = {};
#pragma unroll
    for (int kt = 0; kt < 3; ++kt) {
      v16h a = load_frag(hA, CDIM, mt * 16, kt * 32, lane);
      v16h w = load_frag(qkv_wh, CDIM, nt * 16, kt * 32, lane);
      acc = wmma_step(a, w, acc);
    }
    const int col = nt * 16 + (lane & 15);          // [0,96)
    const int hh = col / HD, d = col & (HD - 1);
    const float bias = qkv_b[col];
#pragma unroll
    for (int i = 0; i < 8; ++i) {
      const int row = mt * 16 + i + ((lane >> 4) << 3);
      qS[hh * NPAD * HD + row * HD + d] = (_Float16)((acc[i] + bias) * QSCALE);
    }
  }
  // k tiles: output cols [96,192)
  for (int job = wave; job < 24; job += 8) {
    const int mt = job / 6, nt = job % 6;
    v8f acc = {};
#pragma unroll
    for (int kt = 0; kt < 3; ++kt) {
      v16h a = load_frag(hA, CDIM, mt * 16, kt * 32, lane);
      v16h w = load_frag(qkv_wh + (size_t)CDIM * CDIM, CDIM, nt * 16, kt * 32, lane);
      acc = wmma_step(a, w, acc);
    }
    const int col = nt * 16 + (lane & 15);          // [0,96) within k block
    const int hh = col / HD, d = col & (HD - 1);
    const float bias = qkv_b[CDIM + col];
#pragma unroll
    for (int i = 0; i < 8; ++i) {
      const int row = mt * 16 + i + ((lane >> 4) << 3);
      kS[hh * NPAD * HD + row * HD + d] = (_Float16)(acc[i] + bias);
    }
  }
  // v tiles: output cols [192,288), stored transposed per head
  for (int job = wave; job < 24; job += 8) {
    const int mt = job / 6, nt = job % 6;
    v8f acc = {};
#pragma unroll
    for (int kt = 0; kt < 3; ++kt) {
      v16h a = load_frag(hA, CDIM, mt * 16, kt * 32, lane);
      v16h w = load_frag(qkv_wh + (size_t)2 * CDIM * CDIM, CDIM, nt * 16, kt * 32, lane);
      acc = wmma_step(a, w, acc);
    }
    const int col = nt * 16 + (lane & 15);          // [0,96) within v block
    const int hh = col / HD, d = col & (HD - 1);
    const float bias = qkv_b[2 * CDIM + col];
#pragma unroll
    for (int i = 0; i < 8; ++i) {
      const int row = mt * 16 + i + ((lane >> 4) << 3);
      vT[hh * HD * NPAD + d * NPAD + row] = (_Float16)(acc[i] + bias);
    }
  }
  __syncthreads();

  // ---- Phase 3..5 per head (SP buffer reused; keeps LDS under 64 KB) ----
  const float* maskw = attn_mask + (size_t)nw * NTOK * NTOK;
  for (int h = 0; h < NHEAD; ++h) {
    // S = q @ k^T + rel_pos_bias + shift_mask  (f16 logits)
    for (int job = wave; job < 16; job += 8) {
      const int mt = job >> 2, nt = job & 3;
      v8f acc = {};
      v16h a = load_frag(qS + h * NPAD * HD, HD, mt * 16, 0, lane);
      v16h bb = load_frag(kS + h * NPAD * HD, HD, nt * 16, 0, lane);
      acc = wmma_step(a, bb, acc);
      const int colm = nt * 16 + (lane & 15);
#pragma unroll
      for (int i = 0; i < 8; ++i) {
        const int row = mt * 16 + i + ((lane >> 4) << 3);
        float val;
        if (row < NTOK && colm < NTOK) {
          const int idx = rel_idx[row * NTOK + colm];
          val = acc[i] + rpb[idx * NHEAD + h] + maskw[row * NTOK + colm];
        } else {
          val = NEGBIG;
        }
        SP[row * NPAD + colm] = (_Float16)val;
      }
    }
    __syncthreads();

    // softmax over 64 cols (padded cols ~ exp(-30000) = 0), in place f16
    if (tid < NPAD) {
      _Float16* sr = SP + tid * NPAD;
      float mx = -3.0e38f;
      for (int c = 0; c < NPAD; ++c) mx = fmaxf(mx, (float)sr[c]);
      float sum = 0.f;
      for (int c = 0; c < NPAD; ++c) sum += __expf((float)sr[c] - mx);
      const float inv = 1.0f / sum;
      for (int c = 0; c < NPAD; ++c) {
        const float e = __expf((float)sr[c] - mx) * inv;
        sr[c] = (_Float16)e;
      }
    }
    __syncthreads();

    // O[:, h*32 : h*32+32] = P(64x64) @ v(64x32)   (B from vT, row-major 32x64)
    for (int job = wave; job < 8; job += 8) {
      const int mt = job >> 1, nt = job & 1;
      v8f acc = {};
#pragma unroll
      for (int kt = 0; kt < 2; ++kt) {
        v16h a = load_frag(SP, NPAD, mt * 16, kt * 32, lane);
        v16h bb = load_frag(vT + h * HD * NPAD, NPAD, nt * 16, kt * 32, lane);
        acc = wmma_step(a, bb, acc);
      }
      const int dcol = nt * 16 + (lane & 15);
#pragma unroll
      for (int i = 0; i < 8; ++i) {
        const int row = mt * 16 + i + ((lane >> 4) << 3);
        O[row * CDIM + h * HD + dcol] = (_Float16)acc[i];
      }
    }
    __syncthreads();
  }

  // ---- Phase 6: proj + residual with reverse shift scatter ----
  for (int job = wave; job < 24; job += 8) {
    const int mt = job / 6, nt = job % 6;
    v8f acc = {};
#pragma unroll
    for (int kt = 0; kt < 3; ++kt) {
      v16h a = load_frag(O, CDIM, mt * 16, kt * 32, lane);
      v16h w = load_frag(proj_wh, CDIM, nt * 16, kt * 32, lane);
      acc = wmma_step(a, w, acc);
    }
    const int col = nt * 16 + (lane & 15);
    const float bias = proj_b[col];
#pragma unroll
    for (int i = 0; i < 8; ++i) {
      const int row = mt * 16 + i + ((lane >> 4) << 3);
      if (row < NTOK) {
        const int ty = row / 7, tx = row % 7;
        const int sr = (wy * 7 + ty + SHIFTV) % 56;
        const int sc = (wx * 7 + tx + SHIFTV) % 56;
        const size_t gi = ((size_t)b * HW + sr * 56 + sc) * CDIM + col;
        out[gi] = x[gi] + acc[i] + bias;
      }
    }
  }
}

// ---------------- Kernel 2: LN2 + MLP (GELU exact) + residual, in place on out ----------------

__global__ __launch_bounds__(256)
void swin_mlp_kernel(float* __restrict__ xo,          // d_out, holds x1, overwritten with final
                     const float* __restrict__ n2w,
                     const float* __restrict__ n2b,
                     const float* __restrict__ fc1_b,  // (384)
                     const float* __restrict__ fc2_b,  // (96)
                     const _Float16* __restrict__ fc1_wh,   // (384,96) f16
                     const _Float16* __restrict__ fc2_wh) { // (96,384) f16
  extern __shared__ char smem[];
  _Float16* A  = (_Float16*)smem;        // 64 x 96
  _Float16* Hh = A + NPAD * CDIM;        // 64 x 384

  const size_t base = (size_t)blockIdx.x * NPAD;   // token row base
  const int tid = threadIdx.x;
  const int wave = tid >> 5, lane = tid & 31;

  // LN2
  if (tid < NPAD) {
    const float* px = xo + (base + tid) * CDIM;
    float mu = 0.f;
    for (int c = 0; c < CDIM; ++c) mu += px[c];
    mu *= (1.0f / CDIM);
    float var = 0.f;
    for (int c = 0; c < CDIM; ++c) { float d = px[c] - mu; var += d * d; }
    var *= (1.0f / CDIM);
    const float inv = rsqrtf(var + 1e-5f);
    for (int c = 0; c < CDIM; ++c)
      A[tid * CDIM + c] = (_Float16)((px[c] - mu) * inv * n2w[c] + n2b[c]);
  }
  __syncthreads();

  // H = gelu(A @ fc1^T + b1)   (64x384, K=96)
  for (int job = wave; job < 4 * 24; job += 8) {
    const int mt = job / 24, nt = job % 24;
    v8f acc = {};
#pragma unroll
    for (int kt = 0; kt < 3; ++kt) {
      v16h a = load_frag(A, CDIM, mt * 16, kt * 32, lane);
      v16h w = load_frag(fc1_wh, CDIM, nt * 16, kt * 32, lane);
      acc = wmma_step(a, w, acc);
    }
    const int col = nt * 16 + (lane & 15);
    const float bias = fc1_b[col];
#pragma unroll
    for (int i = 0; i < 8; ++i) {
      const int row = mt * 16 + i + ((lane >> 4) << 3);
      float v = acc[i] + bias;
      v = 0.5f * v * (1.0f + erff(v * 0.70710678118654752f));   // exact GELU
      Hh[row * MLPD + col] = (_Float16)v;
    }
  }
  __syncthreads();

  // out = x1 + H @ fc2^T + b2   (64x96, K=384)
  for (int job = wave; job < 24; job += 8) {
    const int mt = job / 6, nt = job % 6;
    v8f acc = {};
#pragma unroll
    for (int kt = 0; kt < 12; ++kt) {
      v16h a = load_frag(Hh, MLPD, mt * 16, kt * 32, lane);
      v16h w = load_frag(fc2_wh, MLPD, nt * 16, kt * 32, lane);
      acc = wmma_step(a, w, acc);
    }
    const int col = nt * 16 + (lane & 15);
    const float bias = fc2_b[col];
#pragma unroll
    for (int i = 0; i < 8; ++i) {
      const int row = mt * 16 + i + ((lane >> 4) << 3);
      const size_t gi = (base + row) * CDIM + col;
      xo[gi] = xo[gi] + acc[i] + bias;   // read residual then overwrite (single owner)
    }
  }
}

// ---------------- launch ----------------

extern "C" void kernel_launch(void* const* d_in, const int* in_sizes, int n_in,
                              void* d_out, int out_size, void* d_ws, size_t ws_size,
                              hipStream_t stream) {
  (void)in_sizes; (void)n_in; (void)out_size; (void)ws_size;
  const float* x         = (const float*)d_in[0];
  const float* attn_mask = (const float*)d_in[1];
  const int*   rel_idx   = (const int*)  d_in[2];
  const float* qkv_w     = (const float*)d_in[3];
  const float* qkv_b     = (const float*)d_in[4];
  const float* proj_w    = (const float*)d_in[5];
  const float* proj_b    = (const float*)d_in[6];
  const float* rpb       = (const float*)d_in[7];
  const float* n1w       = (const float*)d_in[8];
  const float* n1b       = (const float*)d_in[9];
  const float* n2w       = (const float*)d_in[10];
  const float* n2b       = (const float*)d_in[11];
  const float* fc1_w     = (const float*)d_in[12];
  const float* fc1_b     = (const float*)d_in[13];
  const float* fc2_w     = (const float*)d_in[14];
  const float* fc2_b     = (const float*)d_in[15];
  float* out = (float*)d_out;

  _Float16* wsh     = (_Float16*)d_ws;
  _Float16* qkv_wh  = wsh;                 // 288*96  = 27648
  _Float16* proj_wh = wsh + 27648;         // 96*96   = 9216
  _Float16* fc1_wh  = wsh + 36864;         // 384*96  = 36864
  _Float16* fc2_wh  = wsh + 73728;         // 96*384  = 36864

  cvt_f32_to_f16<<<(27648 + 255) / 256, 256, 0, stream>>>(qkv_w,  qkv_wh,  27648);
  cvt_f32_to_f16<<<(9216  + 255) / 256, 256, 0, stream>>>(proj_w, proj_wh, 9216);
  cvt_f32_to_f16<<<(36864 + 255) / 256, 256, 0, stream>>>(fc1_w,  fc1_wh,  36864);
  cvt_f32_to_f16<<<(36864 + 255) / 256, 256, 0, stream>>>(fc2_w,  fc2_wh,  36864);

  // LDS: hA/O 12288 + q 12288 + k 12288 + vT 12288 + SP 8192 = 57344 B
  swin_attn_kernel<<<NB * NWIN, 256, 57344, stream>>>(
      x, attn_mask, rel_idx, qkv_b, proj_b, rpb, n1w, n1b, qkv_wh, proj_wh, out);

  // LDS: A 12288 + H 49152 = 61440 B
  swin_mlp_kernel<<<(NB * HW) / NPAD, 256, 61440, stream>>>(
      out, n2w, n2b, fc1_b, fc2_b, fc1_wh, fc2_wh);
}
